// GATE19a_44598940401932
// MI455X (gfx1250) — matrix operand order repaired and checked by reading, not compiled, not run on hardware
//
#include <hip/hip_runtime.h>
#include <hip/hip_bf16.h>

// ---------------------------------------------------------------------------
// GNN MetaLayer forward for MI455X (gfx1250, wave32).
// GEMMs: v_wmma_f32_16x16x32_f16. Block = 256 threads = 8 waves = 8 M-tiles
// (128 rows) x 64 cols. B panel (32 x 64) staged in LDS in fragment layout,
// double-buffered; each wave reads fragments with ds_load_b128.
// Shape contracts: N % 64 == 0; K % 32 in {0,16}; M row-clamped/guarded.
// Assumptions: params dict flattened per-leaf in insertion order (d_in[5..48]),
// num_graphs == 256, ws_size >= ~1.2 GB.
// ---------------------------------------------------------------------------

typedef __attribute__((ext_vector_type(16))) _Float16 v16h;
typedef __attribute__((ext_vector_type(2)))  _Float16 h2v;
typedef __attribute__((ext_vector_type(8)))  float    v8f;

#define LRELU_SLOPE 0.2f
#define BN_EPS 1e-5f

// v_cvt_pk_rtz_f16_f32: pack two f32 -> two f16 in one VALU op.
__device__ __forceinline__ h2v cvt_pk(float x, float y) {
    return __builtin_bit_cast(h2v, __builtin_amdgcn_cvt_pkrtz(x, y));
}

// ---------------------------------------------------------------------------
// Generic fills
// ---------------------------------------------------------------------------
__global__ void k_fill_f32(float* __restrict__ p, size_t n, float v) {
    size_t i = (size_t)blockIdx.x * blockDim.x + threadIdx.x;
    size_t st = (size_t)gridDim.x * blockDim.x;
    for (; i < n; i += st) p[i] = v;
}
__global__ void k_fill_i32(int* __restrict__ p, size_t n, int v) {
    size_t i = (size_t)blockIdx.x * blockDim.x + threadIdx.x;
    size_t st = (size_t)gridDim.x * blockDim.x;
    for (; i < n; i += st) p[i] = v;
}

// ---------------------------------------------------------------------------
// WMMA fragment helpers. Layouts per ISA 7.12.2:
//   A (16x32 f16): lane = half*16 + m ; elem e: K = e + (e>=8?8:0) + half*8
//     -> two contiguous 8-float runs at kk+half*8 and kk+16+half*8
//   B (32x16 f16): lane = half*16 + n ; elem e: K = e + (e>=8?8:0) + half*8
//   C/D (f32):     VGPR r, lane = half*16 + n : M = r + half*8
// LDS B-panel layout (per 64-col step): halves index
//   idx = nt*512 + lane*16 + e   (nt = n/16, lane = ((k/8)%2)*16 + n%16, e = k%8 + (k/16)*8)
// so a lane's fragment for tile nt is 32 contiguous bytes.
// ---------------------------------------------------------------------------
__device__ __forceinline__ v16h a_frag_full(const float* __restrict__ seg,
                                            int kk, int half) {
    const float4* p1 = (const float4*)(seg + kk + half * 8);
    const float4* p2 = (const float4*)(seg + kk + 16 + half * 8);
    float4 f0 = p1[0], f1 = p1[1], f2 = p2[0], f3 = p2[1];
    v16h a; h2v t;
    t = cvt_pk(f0.x, f0.y); a[0] = t[0];  a[1] = t[1];
    t = cvt_pk(f0.z, f0.w); a[2] = t[0];  a[3] = t[1];
    t = cvt_pk(f1.x, f1.y); a[4] = t[0];  a[5] = t[1];
    t = cvt_pk(f1.z, f1.w); a[6] = t[0];  a[7] = t[1];
    t = cvt_pk(f2.x, f2.y); a[8] = t[0];  a[9] = t[1];
    t = cvt_pk(f2.z, f2.w); a[10] = t[0]; a[11] = t[1];
    t = cvt_pk(f3.x, f3.y); a[12] = t[0]; a[13] = t[1];
    t = cvt_pk(f3.z, f3.w); a[14] = t[0]; a[15] = t[1];
    return a;
}
// 16-wide K tail: upper run zero.
__device__ __forceinline__ v16h a_frag_half(const float* __restrict__ seg,
                                            int kk, int half) {
    const float4* p1 = (const float4*)(seg + kk + half * 8);
    float4 f0 = p1[0], f1 = p1[1];
    v16h a; h2v t;
    t = cvt_pk(f0.x, f0.y); a[0] = t[0]; a[1] = t[1];
    t = cvt_pk(f0.z, f0.w); a[2] = t[0]; a[3] = t[1];
    t = cvt_pk(f1.x, f1.y); a[4] = t[0]; a[5] = t[1];
    t = cvt_pk(f1.z, f1.w); a[6] = t[0]; a[7] = t[1];
#pragma unroll
    for (int e = 8; e < 16; ++e) a[e] = (_Float16)0.f;
    return a;
}

// LDS half-index for B fragment element (k0l, n0).
__device__ __forceinline__ int b_lds_idx(int k0l, int n0) {
    return ((n0 >> 4) << 9) + ((((k0l >> 3) & 1) << 4) + (n0 & 15)) * 16 + (k0l & 7) + ((k0l >> 4) << 3);
}

// Cooperative B-panel stage, branch-free (full 32 K-rows).
// Per-thread: fixed column n0 and base row kl0; tasks step rows by 8.
// wb = W + nbase + n0 + kl0*ldw (hoisted by caller); offset adds kg*ldw.
__device__ __forceinline__ void stage_b32(_Float16* dst, const float* __restrict__ wb,
                                          int ldw, int kg, int n0, int kl0) {
#pragma unroll
    for (int it = 0; it < 4; ++it) {
        const float* wp = wb + (size_t)(kg + it * 8) * ldw;
        h2v t = cvt_pk(wp[0], wp[ldw]);
        *(h2v*)(dst + b_lds_idx(kl0 + it * 8, n0)) = t;
    }
}
// Guarded variant for the final (K%32==16) stage only.
__device__ __forceinline__ void stage_b_tail(_Float16* dst, const float* __restrict__ wb,
                                             int ldw, int kg, int n0, int kl0, int kwidth) {
#pragma unroll
    for (int it = 0; it < 4; ++it) {
        int k0l = kl0 + it * 8;
        float w0 = 0.f, w1 = 0.f;
        if (k0l < kwidth) {
            const float* wp = wb + (size_t)(kg + it * 8) * ldw;
            w0 = wp[0]; w1 = wp[ldw];
        }
        h2v t = cvt_pk(w0, w1);
        *(h2v*)(dst + b_lds_idx(k0l, n0)) = t;
    }
}

// 4 WMMAs against the staged panel.
__device__ __forceinline__ void wmma4(v16h a, const _Float16* lb, int lane,
                                      v8f acc[4]) {
#pragma unroll
    for (int nt = 0; nt < 4; ++nt) {
        v16h b = *(const v16h*)(lb + nt * 512 + lane * 16);
        acc[nt] = __builtin_amdgcn_wmma_f32_16x16x32_f16(false, a, false, b,
                                                         (short)0, acc[nt], false, false);
    }
}

__device__ __forceinline__ void wmma_epilogue(
    v8f acc[4], const float* __restrict__ bias, float* __restrict__ C, int ldc,
    int mbase, int nbase, int lr, int half, int M, int relu) {
#pragma unroll
    for (int nt = 0; nt < 4; ++nt) {
        int bcol = nbase + nt * 16 + lr;
        float bs = bias ? bias[bcol] : 0.f;
#pragma unroll
        for (int r = 0; r < 8; ++r) {
            int mrow = mbase + r + half * 8;
            if (mrow < M) {
                float v = acc[nt][r] + bs;
                if (relu) v = fmaxf(v, 0.f);
                C[(size_t)mrow * ldc + bcol] = v;
            }
        }
    }
}

// ---------------------------------------------------------------------------
// Plain WMMA GEMM: C[M,N] = act(A[M,K] @ W[K,N] + bias), K % 32 == 0
// grid = (N/64, ceil(M/128)), block = 256 (8 waves -> 128x64 tile)
// ---------------------------------------------------------------------------
__global__ void __launch_bounds__(256)
k_gemm(const float* __restrict__ A, int lda,
       const float* __restrict__ W, int ldw,
       const float* __restrict__ bias,
       float* __restrict__ C, int ldc,
       int M, int K, int relu) {
    __shared__ _Float16 lbuf[2][2048];
    const int tid  = threadIdx.x;
    const int lane = tid & 31;
    const int wave = tid >> 5;
    const int half = lane >> 4;
    const int lr   = lane & 15;
    const int mbase = blockIdx.y * 128 + wave * 16;
    const int nbase = blockIdx.x * 64;
    const int arow  = (mbase + lr < M) ? (mbase + lr) : (M - 1);   // clamp
    const float* aRow = A + (size_t)arow * lda;
    // staging coords: fixed column, base row per thread
    const int n0  = tid & 63;
    const int kl0 = (tid >> 6) << 1;
    const float* wb = W + (size_t)kl0 * ldw + (nbase + n0);

    stage_b32(&lbuf[0][0], wb, ldw, 0, n0, kl0);
    __syncthreads();

    v8f z = {};
    v8f acc[4] = {z, z, z, z};
    for (int k0 = 0; k0 < K; k0 += 32) {
        const int p = (k0 >> 5) & 1;
        if (k0 + 32 < K)
            stage_b32(&lbuf[1 - p][0], wb, ldw, k0 + 32, n0, kl0);
        __builtin_prefetch(aRow + k0 + 128, 0, 1);   // speculative, safe OOB
        v16h a = a_frag_full(aRow, k0, half);
        wmma4(a, &lbuf[p][0], lane, acc);
        __syncthreads();
    }
    wmma_epilogue(acc, bias, C, ldc, mbase, nbase, lr, half, M, relu);
}

// ---------------------------------------------------------------------------
// Gather-concat WMMA GEMM: A[r] = [ H[src[r]] | H[tgt[r]] | aux[r] ]
// K = 2*c01 + caux, caux % 32 in {0,16}.
// (H is L2-resident: 51 MB << 192 MB, so gathers stay on-die.)
// ---------------------------------------------------------------------------
__global__ void __launch_bounds__(256)
k_gemm_cat3(const float* __restrict__ H, int c01,
            const float* __restrict__ aux, int caux,
            const int* __restrict__ src, const int* __restrict__ tgt,
            const float* __restrict__ W, int ldw,
            const float* __restrict__ bias,
            float* __restrict__ C, int ldc, int M, int relu) {
    __shared__ _Float16 lbuf[2][2048];
    const int tid  = threadIdx.x;
    const int lane = tid & 31;
    const int wave = tid >> 5;
    const int half = lane >> 4;
    const int lr   = lane & 15;
    const int mbase = blockIdx.y * 128 + wave * 16;
    const int nbase = blockIdx.x * 64;
    const int arow  = (mbase + lr < M) ? (mbase + lr) : (M - 1);   // clamp
    const int K     = 2 * c01 + caux;
    const int s = src[arow];
    const int t = tgt[arow];
    const float* seg0 = H + (size_t)s * c01;
    const float* seg1 = H + (size_t)t * c01;
    const float* seg2 = aux + (size_t)arow * caux;
    const int n0  = tid & 63;
    const int kl0 = (tid >> 6) << 1;
    const float* wb = W + (size_t)kl0 * ldw + (nbase + n0);

    stage_b32(&lbuf[0][0], wb, ldw, 0, n0, kl0);
    __syncthreads();

    v8f z = {};
    v8f acc[4] = {z, z, z, z};
    int kg = 0;
    // one K-step: stage next panel (branch-free unless it's the 16-wide tail),
    // build A fragment from `seg`, run 4 WMMAs, barrier.
    auto step = [&](const float* seg, int kk) {
        const int p = (kg >> 5) & 1;
        const int kn = kg + 32;                 // next staged block base
        if (kn < K) {
            if (kn + 32 <= K) stage_b32(&lbuf[1 - p][0], wb, ldw, kn, n0, kl0);
            else              stage_b_tail(&lbuf[1 - p][0], wb, ldw, kn, n0, kl0, K - kn);
        }
        v16h a = a_frag_full(seg, kk, half);
        wmma4(a, &lbuf[p][0], lane, acc);
        __syncthreads();
        kg += 32;
    };
    for (int kk = 0; kk < c01; kk += 32) step(seg0, kk);
    for (int kk = 0; kk < c01; kk += 32) step(seg1, kk);
    const int fullc = caux & ~31;
    for (int kk = 0; kk < fullc; kk += 32) step(seg2, kk);
    if (caux & 16) {
        const int p = (kg >> 5) & 1;
        v16h a = a_frag_half(seg2, fullc, half);
        wmma4(a, &lbuf[p][0], lane, acc);
    }
    wmma_epilogue(acc, bias, C, ldc, mbase, nbase, lr, half, M, relu);
}

// ---------------------------------------------------------------------------
// Self-loop attr: count incoming edges, scatter-sum edge features, divide.
// ---------------------------------------------------------------------------
__global__ void k_edge_cnt(const int* __restrict__ tgt, float* __restrict__ cnt, int E) {
    int j = blockIdx.x * blockDim.x + threadIdx.x;
    if (j < E) atomicAdd(&cnt[tgt[j]], 1.f);
}
__global__ void k_scatter_rows(const float* __restrict__ X, const int* __restrict__ tgt,
                               float* __restrict__ sum, int E, int D) {
    size_t i = (size_t)blockIdx.x * blockDim.x + threadIdx.x;
    size_t tot = (size_t)E * D;
    size_t st = (size_t)gridDim.x * blockDim.x;
    for (; i < tot; i += st) {
        int j = (int)(i / D), d = (int)(i % D);
        atomicAdd(&sum[(size_t)tgt[j] * D + d], X[i]);
    }
}
__global__ void k_rows_div(float* __restrict__ sum, const float* __restrict__ cnt,
                           int N, int D) {
    size_t i = (size_t)blockIdx.x * blockDim.x + threadIdx.x;
    size_t tot = (size_t)N * D;
    size_t st = (size_t)gridDim.x * blockDim.x;
    for (; i < tot; i += st) sum[i] /= fmaxf(cnt[i / D], 1.f);
}

// ---------------------------------------------------------------------------
// GATv2 attention. j in [0, E) = real edges, j in [E, E+N) = self-loops.
// Channel layout: 256 = [H=4][C=64]; att flattened (4,64) matches channel idx.
// ---------------------------------------------------------------------------
__device__ __forceinline__ int enc_ord(float v) {
    int e = __float_as_int(v);
    return e ^ ((e >> 31) & 0x7fffffff);
}
__device__ __forceinline__ float dec_ord(int e) {
    return __int_as_float(e ^ ((e >> 31) & 0x7fffffff));
}

__global__ void k_att_logit(const float* __restrict__ xl, const float* __restrict__ xr,
                            const float* __restrict__ eeE, const float* __restrict__ eeN,
                            const int* __restrict__ src, const int* __restrict__ tgt,
                            const float* __restrict__ att,
                            float* __restrict__ logit, int* __restrict__ mmax,
                            int E, int N) {
    int j = blockIdx.x * (blockDim.x >> 5) + (threadIdx.x >> 5);
    if (j >= E + N) return;
    int lane = threadIdx.x & 31;
    int s, t; const float* ee;
    if (j < E) { s = src[j]; t = tgt[j]; ee = eeE + (size_t)j * 256; }
    else       { s = t = j - E;          ee = eeN + (size_t)(j - E) * 256; }
    float acch[4] = {0.f, 0.f, 0.f, 0.f};
#pragma unroll
    for (int i = 0; i < 8; ++i) {
        int c = lane + i * 32;
        float zv = xl[(size_t)s * 256 + c] + xr[(size_t)t * 256 + c] + ee[c];
        zv = (zv > 0.f) ? zv : LRELU_SLOPE * zv;
        acch[c >> 6] += zv * att[c];
    }
#pragma unroll
    for (int h = 0; h < 4; ++h) {
        float v = acch[h];
        for (int o = 16; o > 0; o >>= 1) v += __shfl_xor(v, o, 32);
        if (lane == 0) {
            logit[(size_t)j * 4 + h] = v;
            atomicMax(&mmax[t * 4 + h], enc_ord(v));
        }
    }
}

__global__ void k_att_exp(float* __restrict__ logit, const int* __restrict__ mmax,
                          float* __restrict__ den, const int* __restrict__ tgt,
                          int E, int N) {
    int j = blockIdx.x * blockDim.x + threadIdx.x;
    if (j >= E + N) return;
    int t = (j < E) ? tgt[j] : (j - E);
#pragma unroll
    for (int h = 0; h < 4; ++h) {
        float m = dec_ord(mmax[t * 4 + h]);
        float ex = expf(logit[(size_t)j * 4 + h] - m);
        logit[(size_t)j * 4 + h] = ex;           // logit buffer now holds exp
        atomicAdd(&den[t * 4 + h], ex);
    }
}

__global__ void k_att_agg(const float* __restrict__ ex, const float* __restrict__ den,
                          const float* __restrict__ xl,
                          const int* __restrict__ src, const int* __restrict__ tgt,
                          float* __restrict__ agg, int E, int N) {
    int j = blockIdx.x * (blockDim.x >> 5) + (threadIdx.x >> 5);
    if (j >= E + N) return;
    int lane = threadIdx.x & 31;
    int s, t;
    if (j < E) { s = src[j]; t = tgt[j]; } else { s = t = j - E; }
#pragma unroll
    for (int i = 0; i < 8; ++i) {
        int c = lane + i * 32;
        int h = c >> 6;
        float a = ex[(size_t)j * 4 + h] / den[(size_t)t * 4 + h];
        atomicAdd(&agg[(size_t)t * 256 + c], a * xl[(size_t)s * 256 + c]);
    }
}

__global__ void k_bias_relu(float* __restrict__ X, const float* __restrict__ bias,
                            size_t M, int C) {
    size_t i = (size_t)blockIdx.x * blockDim.x + threadIdx.x;
    size_t tot = M * C;
    size_t st = (size_t)gridDim.x * blockDim.x;
    for (; i < tot; i += st) X[i] = fmaxf(X[i] + bias[i % C], 0.f);
}

// ---------------------------------------------------------------------------
// Graph pooling + global concat + BatchNorm
// ---------------------------------------------------------------------------
__global__ void k_pool(const float* __restrict__ X, const int* __restrict__ batch,
                       float* __restrict__ pool, int N, int C) {
    size_t i = (size_t)blockIdx.x * blockDim.x + threadIdx.x;
    size_t tot = (size_t)N * C;
    size_t st = (size_t)gridDim.x * blockDim.x;
    for (; i < tot; i += st) {
        int r = (int)(i / C), c = (int)(i % C);
        atomicAdd(&pool[(size_t)batch[r] * C + c], X[i]);
    }
}
__global__ void k_concat_u(const float* __restrict__ u, const float* __restrict__ pool,
                           float* __restrict__ catu, int G) {
    size_t i = (size_t)blockIdx.x * blockDim.x + threadIdx.x;
    size_t tot = (size_t)G * 640;
    size_t st = (size_t)gridDim.x * blockDim.x;
    for (; i < tot; i += st) {
        int r = (int)(i / 640), c = (int)(i % 640);
        catu[i] = (c < 384) ? u[(size_t)r * 384 + c] : pool[(size_t)r * 256 + (c - 384)];
    }
}
__global__ void k_colstats(const float* __restrict__ X, int M, int C,
                           float* __restrict__ mean, float* __restrict__ var) {
    int c = blockIdx.x;
    float a = 0.f, b = 0.f;
    for (size_t r = threadIdx.x; r < (size_t)M; r += blockDim.x) {
        float v = X[r * C + c];
        a += v; b += v * v;
    }
    __shared__ float s1[256], s2[256];
    s1[threadIdx.x] = a; s2[threadIdx.x] = b;
    __syncthreads();
    for (int st = 128; st > 0; st >>= 1) {
        if ((int)threadIdx.x < st) {
            s1[threadIdx.x] += s1[threadIdx.x + st];
            s2[threadIdx.x] += s2[threadIdx.x + st];
        }
        __syncthreads();
    }
    if (threadIdx.x == 0) {
        float m = s1[0] / (float)M;
        mean[c] = m;
        var[c]  = s2[0] / (float)M - m * m;
    }
}
__global__ void k_bn_apply(float* __restrict__ X, size_t M, int C,
                           const float* __restrict__ mean, const float* __restrict__ var,
                           const float* __restrict__ g, const float* __restrict__ b) {
    size_t i = (size_t)blockIdx.x * blockDim.x + threadIdx.x;
    size_t tot = M * C;
    size_t st = (size_t)gridDim.x * blockDim.x;
    for (; i < tot; i += st) {
        int c = (int)(i % C);
        X[i] = (X[i] - mean[c]) * rsqrtf(var[c] + BN_EPS) * g[c] + b[c];
    }
}

// fc2: [G,64] @ [64,1] + b
__global__ void k_fc2(const float* __restrict__ X, const float* __restrict__ w,
                      const float* __restrict__ b, float* __restrict__ out, int G) {
    int g = blockIdx.x * blockDim.x + threadIdx.x;
    if (g >= G) return;
    float acc = b[0];
#pragma unroll
    for (int k = 0; k < 64; ++k) acc += X[(size_t)g * 64 + k] * w[k];
    out[g] = acc;
}

// ---------------------------------------------------------------------------
// Host-side launch helpers
// ---------------------------------------------------------------------------
static inline unsigned gdiv(size_t n, unsigned b) { return (unsigned)((n + b - 1) / b); }

static inline void gemm(hipStream_t st, const float* A, int lda, const float* W, int ldw,
                        const float* bias, float* C, int ldc, int M, int K, int Nn, int relu) {
    dim3 grid((unsigned)(Nn / 64), gdiv((size_t)M, 128)), blk(256);
    k_gemm<<<grid, blk, 0, st>>>(A, lda, W, ldw, bias, C, ldc, M, K, relu);
}
static inline void gemm_cat3(hipStream_t st, const float* H, int c01, const float* aux, int caux,
                             const int* src, const int* tgt, const float* W, int ldw,
                             const float* bias, float* C, int ldc, int M, int Nn, int relu) {
    dim3 grid((unsigned)(Nn / 64), gdiv((size_t)M, 128)), blk(256);
    k_gemm_cat3<<<grid, blk, 0, st>>>(H, c01, aux, caux, src, tgt, W, ldw, bias, C, ldc, M, relu);
}

extern "C" void kernel_launch(void* const* d_in, const int* in_sizes, int n_in,
                              void* d_out, int out_size, void* d_ws, size_t ws_size,
                              hipStream_t stream) {
    const int N = in_sizes[0] / 128;       // 50000 (multiple of 16)
    const int E = in_sizes[1] / 16;        // 400000 (multiple of 128)
    const int G = 256;                     // num_graphs (fixed by harness)

    const float* x   = (const float*)d_in[0];
    const float* ea  = (const float*)d_in[1];
    const int*   src = (const int*)d_in[2];
    const int*   tgt = src + E;
    const int*   bat = (const int*)d_in[3];
    #define PF(i) ((const float*)d_in[i])
    // params (dict insertion order, leaves flattened):
    // 5 nt_W1 6 nt_b1 7 nt_W2 8 nt_b2 | 9 e1_W1 10 e1_b1 11 e1_W2 12 e1_b2
    // 13 n1_Wl 14 n1_bl 15 n1_Wr 16 n1_br 17 n1_We 18 n1_att 19 n1_bias
    // 20 g1_W1 21 g1_b1 22 g1_W2 23 g1_b2 | 24..29 bn_{n,e,u}_{g,b}
    // 30 e2_W1 31 e2_b1 32 e2_W2 33 e2_b2
    // 34 n2_Wl 35 n2_bl 36 n2_Wr 37 n2_br 38 n2_We 39 n2_att 40 n2_bias
    // 41 g2_W1 42 g2_b1 43 g2_W2 44 g2_b2 | 45 fc1_W 46 fc1_b 47 fc2_W 48 fc2_b

    // ---- workspace layout (floats) ----
    float* base = (float*)d_ws;
    size_t off = 0;
    auto alloc = [&](size_t n) { float* p = base + off; off += n; return p; };
    float* h    = alloc((size_t)N * 256);
    float* tE   = alloc((size_t)E * 128);   // MLP hiddens (node hidden N*256 fits too)
    float* ef   = alloc((size_t)E * 128);   // e (layer1), then e2
    float* xl   = alloc((size_t)N * 256);
    float* xr   = alloc((size_t)N * 256);
    float* eeE  = alloc((size_t)E * 256);
    float* eeN  = alloc((size_t)N * 256);
    float* agg  = alloc((size_t)N * 256);   // GAT accumulator, then hn
    float* lsum = alloc((size_t)N * 128);
    float* cnt  = alloc((size_t)N);
    float* lgt  = alloc((size_t)(E + N) * 4);
    float* den  = alloc((size_t)N * 4);
    int*   mmx  = (int*)alloc((size_t)N * 4);
    float* pool = alloc((size_t)G * 256);
    float* catu = alloc((size_t)G * 640);
    float* tG   = alloc((size_t)G * 384);
    float* u    = alloc((size_t)G * 384);
    float* th   = alloc((size_t)G * 64);
    float* bnm  = alloc(384);
    float* bnv  = alloc(384);
    (void)ws_size; (void)n_in; (void)out_size;

    const dim3 B256(256);
    const unsigned GSTRIDE = 4096;   // cap for grid-stride kernels

    // ================= node transform MLP =================
    gemm(stream, x, 128, PF(5), 256, PF(6), tE, 256, N, 128, 256, 1);
    gemm(stream, tE, 256, PF(7), 256, PF(8), h, 256, N, 256, 256, 0);

    for (int layer = 0; layer < 2; ++layer) {
        const float *Wl, *bl, *Wr, *br, *We, *att, *nbias;
        const float *eW1, *eb1, *eW2, *eb2, *gW1, *gb1, *gW2, *gb2;
        const float* hin = (layer == 0) ? h : agg;   // layer2 node input = hn (in agg)
        int eH;                                      // edge-MLP hidden width
        const float* aux; int caux;
        if (layer == 0) {
            eW1 = PF(9);  eb1 = PF(10); eW2 = PF(11); eb2 = PF(12);
            Wl = PF(13); bl = PF(14); Wr = PF(15); br = PF(16);
            We = PF(17); att = PF(18); nbias = PF(19);
            gW1 = PF(20); gb1 = PF(21); gW2 = PF(22); gb2 = PF(23);
            aux = ea; caux = 16; eH = 64;
        } else {
            eW1 = PF(30); eb1 = PF(31); eW2 = PF(32); eb2 = PF(33);
            Wl = PF(34); bl = PF(35); Wr = PF(36); br = PF(37);
            We = PF(38); att = PF(39); nbias = PF(40);
            gW1 = PF(41); gb1 = PF(42); gW2 = PF(43); gb2 = PF(44);
            aux = ef; caux = 128; eH = 128;
        }

        // ---- edge MLP: e = mlp2([h[src] | h[tgt] | aux]) ----
        gemm_cat3(stream, hin, 256, aux, caux, src, tgt, eW1, eH, eb1, tE, eH, E, eH, 1);
        gemm(stream, tE, eH, eW2, 128, eb2, ef, 128, E, eH, 128, 0);

        // ---- GATv2 projections ----
        gemm(stream, hin, 256, Wl, 256, bl, xl, 256, N, 256, 256, 0);
        gemm(stream, hin, 256, Wr, 256, br, xr, 256, N, 256, 256, 0);

        // ---- self-loop attr = segment-mean of e over tgt ----
        k_fill_f32<<<GSTRIDE, B256, 0, stream>>>(lsum, (size_t)N * 128, 0.f);
        k_fill_f32<<<gdiv(N, 256), B256, 0, stream>>>(cnt, (size_t)N, 0.f);
        k_edge_cnt<<<gdiv(E, 256), B256, 0, stream>>>(tgt, cnt, E);
        k_scatter_rows<<<GSTRIDE, B256, 0, stream>>>(ef, tgt, lsum, E, 128);
        k_rows_div<<<GSTRIDE, B256, 0, stream>>>(lsum, cnt, N, 128);

        // ---- ee = e @ We (edges), loop_attr @ We (self-loops) ----
        gemm(stream, ef, 128, We, 256, nullptr, eeE, 256, E, 128, 256, 0);
        gemm(stream, lsum, 128, We, 256, nullptr, eeN, 256, N, 128, 256, 0);

        // ---- attention softmax over incoming edges ----
        k_fill_i32<<<gdiv((size_t)N * 4, 256), B256, 0, stream>>>(mmx, (size_t)N * 4, (int)0x80000000);
        {
            dim3 blk(128);                         // 4 waves/block, wave-per-edge
            dim3 grd(gdiv((size_t)(E + N), 4));
            k_att_logit<<<grd, blk, 0, stream>>>(xl, xr, eeE, eeN, src, tgt, att, lgt, mmx, E, N);
        }
        k_fill_f32<<<gdiv((size_t)N * 4, 256), B256, 0, stream>>>(den, (size_t)N * 4, 0.f);
        k_att_exp<<<gdiv((size_t)(E + N), 256), B256, 0, stream>>>(lgt, mmx, den, tgt, E, N);

        // ---- aggregate: hn[t] += alpha * xl[s]  (hn lives in agg) ----
        k_fill_f32<<<GSTRIDE, B256, 0, stream>>>(agg, (size_t)N * 256, 0.f);
        {
            dim3 blk(128);
            dim3 grd(gdiv((size_t)(E + N), 4));
            k_att_agg<<<grd, blk, 0, stream>>>(lgt, den, xl, src, tgt, agg, E, N);
        }
        k_bias_relu<<<GSTRIDE, B256, 0, stream>>>(agg, nbias, (size_t)N, 256);

        // ---- graph pooling + global MLP ----
        k_fill_f32<<<gdiv((size_t)G * 256, 256), B256, 0, stream>>>(pool, (size_t)G * 256, 0.f);
        k_pool<<<GSTRIDE, B256, 0, stream>>>(agg, bat, pool, N, 256);
        if (layer == 0)
            k_fill_f32<<<gdiv((size_t)G * 384, 256), B256, 0, stream>>>(u, (size_t)G * 384, 0.f);
        k_concat_u<<<gdiv((size_t)G * 640, 256), B256, 0, stream>>>(u, pool, catu, G);
        gemm(stream, catu, 640, gW1, 384, gb1, tG, 384, G, 640, 384, 1);
        gemm(stream, tG, 384, gW2, 384, gb2, u, 384, G, 384, 384, 0);

        // ---- batch-norms (after MetaLayer 1 only) ----
        if (layer == 0) {
            k_colstats<<<256, B256, 0, stream>>>(agg, N, 256, bnm, bnv);
            k_bn_apply<<<GSTRIDE, B256, 0, stream>>>(agg, (size_t)N, 256, bnm, bnv, PF(24), PF(25));
            k_colstats<<<128, B256, 0, stream>>>(ef, E, 128, bnm, bnv);
            k_bn_apply<<<GSTRIDE, B256, 0, stream>>>(ef, (size_t)E, 128, bnm, bnv, PF(26), PF(27));
            k_colstats<<<384, B256, 0, stream>>>(u, G, 384, bnm, bnv);
            k_bn_apply<<<gdiv((size_t)G * 384, 256), B256, 0, stream>>>(u, (size_t)G, 384, bnm, bnv, PF(28), PF(29));
        }
    }

    // ================= head =================
    gemm(stream, u, 384, PF(45), 64, PF(46), th, 64, G, 384, 64, 1);
    k_fc2<<<gdiv(G, 64), dim3(64), 0, stream>>>(th, PF(47), PF(48), (float*)d_out, G);
    #undef PF
}
